// TernaryBlock_67619965108848
// MI455X (gfx1250) — matrix operand order repaired and checked by reading, not compiled, not run on hardware
//
#include <hip/hip_runtime.h>

// ---------------------------------------------------------------------------
// Ternary inverted-residual block for MI455X (gfx1250, wave32, WMMA).
//
// Pipeline (all on `stream`, intermediates in d_ws, f16 staging so the
// 115MB tensors stay L2-resident on the 192MB L2):
//   1. ternary_stats x3      : delta=0.7*mean|w|, alpha = mean |w| over threshold
//   2. pack ternary weights  : {-1,0,1} as f16 (alpha applied to f32 accum)
//   3. gemm_expand           : 1x1 conv as GEMM (M=100352,K=96,N=576), v_wmma f16
//   4. channel_stats + bn_finalize (BN1)
//   5. dwconv 3x3            : LDS-tiled, BN1+ReLU fused on load
//   6. channel_stats + bn_finalize (BN2)
//   7. gemm_project          : BN2+ReLU fused into A-tile load, double-buffered
//                              K-loop, async B-tile staging, v_wmma f16
//   8. channel_stats + bn_finalize (BN3)
//   9. bn_residual           : out = BN3(y3) + x   (fp32)
// ---------------------------------------------------------------------------

typedef __attribute__((ext_vector_type(16))) _Float16 v16h;
typedef __attribute__((ext_vector_type(8)))  _Float16 v8h;
typedef __attribute__((ext_vector_type(8)))  float    v8f;

constexpr int kBn  = 32;        // batch
constexpr int kC   = 96;        // in/out channels
constexpr int kHW  = 3136;      // 56*56
constexpr int kP   = 576;       // expanded planes
constexpr int kM   = kBn * kHW; // 100352 GEMM rows

// ---- CDNA5 async global->LDS path (ASYNCcnt), guarded so both toolchains
// ---- compile; falls back to plain ds stores if the builtin is absent.
#if defined(__HIP_DEVICE_COMPILE__) && defined(__gfx1250__) && \
    __has_builtin(__builtin_amdgcn_global_load_async_to_lds_b128)
#define HAVE_ASYNC_LDS 1
#else
#define HAVE_ASYNC_LDS 0
#endif

#if HAVE_ASYNC_LDS
// Builtin signature (from hipcc diagnostic): param0 is
// 'int __vector(4) __device__ *'  == vector_size(16) int, address_space(1).
typedef __attribute__((__vector_size__(16))) int v4i_b;
typedef __attribute__((address_space(1))) v4i_b* gv4p;  // global src
typedef __attribute__((address_space(3))) v4i_b* lv4p;  // LDS dst
static __device__ inline void async_cp16(void* lds, const void* g) {
  __builtin_amdgcn_global_load_async_to_lds_b128((gv4p)g, (lv4p)lds, 0, 0);
}
#endif

static __device__ inline void wait_async_all() {
#if HAVE_ASYNC_LDS
#if __has_builtin(__builtin_amdgcn_s_wait_asynccnt)
  __builtin_amdgcn_s_wait_asynccnt(0);
#else
  asm volatile("s_wait_asynccnt 0x0" ::: "memory");
#endif
#endif
}

static __device__ inline v16h combine8(v8h lo, v8h hi) {
  v16h r;
#pragma unroll
  for (int j = 0; j < 8; ++j) { r[j] = lo[j]; r[j + 8] = hi[j]; }
  return r;
}

// ---------------------------------------------------------------------------
// Ternarization statistics: out[0]=delta, out[1]=alpha   (single block)
// ---------------------------------------------------------------------------
__global__ __launch_bounds__(256)
void ternary_stats(const float* __restrict__ w, int n, float* __restrict__ out) {
  __shared__ float rs[256], rq[256];
  __shared__ float bcast;
  const int t = threadIdx.x;
  float sa = 0.f;
  for (int i = t; i < n; i += 256) sa += fabsf(w[i]);
  rs[t] = sa; __syncthreads();
  for (int off = 128; off; off >>= 1) { if (t < off) rs[t] += rs[t + off]; __syncthreads(); }
  if (!t) bcast = 0.7f * rs[0] / (float)n;
  __syncthreads();
  const float delta = bcast;
  float sm = 0.f, cnt = 0.f;
  for (int i = t; i < n; i += 256) {
    float a = fabsf(w[i]);
    if (a > delta) { sm += a; cnt += 1.f; }
  }
  rs[t] = sm; rq[t] = cnt; __syncthreads();
  for (int off = 128; off; off >>= 1) {
    if (t < off) { rs[t] += rs[t + off]; rq[t] += rq[t + off]; }
    __syncthreads();
  }
  if (!t) { out[0] = delta; out[1] = rs[0] / fmaxf(rq[0], 1.f); }
}

// pack ternary sign matrix {-1,0,1} to f16 (exactly representable)
__global__ __launch_bounds__(256)
void pack_tern_f16(const float* __restrict__ w, _Float16* __restrict__ dst,
                   int n, const float* __restrict__ da) {
  int i = blockIdx.x * 256 + threadIdx.x;
  if (i < n) {
    float v = w[i];
    float tn = (fabsf(v) > da[0]) ? (v > 0.f ? 1.f : -1.f) : 0.f;
    dst[i] = (_Float16)tn;
  }
}

// pack alpha-scaled ternary weights to f32 (for the scalar depthwise conv)
__global__ __launch_bounds__(256)
void pack_tern_f32s(const float* __restrict__ w, float* __restrict__ dst,
                    int n, const float* __restrict__ da) {
  int i = blockIdx.x * 256 + threadIdx.x;
  if (i < n) {
    float v = w[i];
    float tn = (fabsf(v) > da[0]) ? (v > 0.f ? 1.f : -1.f) : 0.f;
    dst[i] = da[1] * tn;
  }
}

// ---------------------------------------------------------------------------
// GEMM1: y1[n][m] = alpha1 * sum_k x[m][k] * t1[n][k]
//   block: 256 thr = 8 waves (2M x 4N), block tile 64M x 192N, K = 96 in LDS
// ---------------------------------------------------------------------------
__global__ __launch_bounds__(256)
void gemm_expand(const float* __restrict__ x, const _Float16* __restrict__ w1t,
                 const float* __restrict__ stats, _Float16* __restrict__ y1) {
  __shared__ __attribute__((aligned(16))) _Float16 Al[64 * 96];   // [m][k]
  __shared__ __attribute__((aligned(16))) _Float16 Bl[192 * 96];  // [n][k]
  const int t   = threadIdx.x;
  const int m0  = blockIdx.x * 64;
  const int b   = m0 / kHW;          // 64 | 3136 -> whole tile in one batch image
  const int hw0 = m0 % kHW;
  const int n0  = blockIdx.y * 192;
  const float alpha = stats[1];

#if HAVE_ASYNC_LDS
  // B tile is a contiguous 36KB slab of w1t: async DMA it while we convert A.
  for (int idx = t; idx < (192 * 96) / 8; idx += 256)
    async_cp16(&Bl[idx * 8], &w1t[(size_t)n0 * 96 + idx * 8]);
#else
  for (int idx = t; idx < 192 * 96; idx += 256)
    Bl[idx] = w1t[(size_t)(n0 + idx / 96) * 96 + (idx % 96)];
#endif
  for (int idx = t; idx < 64 * 96; idx += 256) {      // A: x (NCHW) f32 -> f16 [m][k]
    int k = idx >> 6, ml = idx & 63;
    Al[ml * 96 + k] = (_Float16)x[(size_t)b * (kC * kHW) + k * kHW + hw0 + ml];
  }
  wait_async_all();
  __syncthreads();

  const int wave = t >> 5, lane = t & 31;
  const int half = lane >> 4, lr = lane & 15;
  const int wm = wave & 1, wn = wave >> 1;   // 2M x 4N waves

  v8f acc[2][3] = {};
#pragma unroll
  for (int kb = 0; kb < 96; kb += 32) {
    v16h af[2], bf[3];
#pragma unroll
    for (int tm = 0; tm < 2; ++tm) {         // A frag: K split 0..7 / 16..23 per half
      int m = wm * 32 + tm * 16 + lr;
      af[tm] = combine8(*(const v8h*)&Al[m * 96 + kb + half * 8],
                        *(const v8h*)&Al[m * 96 + kb + 16 + half * 8]);
    }
#pragma unroll
    for (int tn = 0; tn < 3; ++tn) {         // B frag: K split 0..7 / 16..23 per half
      int n = wn * 48 + tn * 16 + lr;
      bf[tn] = combine8(*(const v8h*)&Bl[n * 96 + kb + half * 8],
                        *(const v8h*)&Bl[n * 96 + kb + 16 + half * 8]);
    }
#pragma unroll
    for (int tm = 0; tm < 2; ++tm)
#pragma unroll
      for (int tn = 0; tn < 3; ++tn)
        acc[tm][tn] = __builtin_amdgcn_wmma_f32_16x16x32_f16(
            false, af[tm], false, bf[tn], (short)0, acc[tm][tn], false, false);
  }

#pragma unroll
  for (int tn = 0; tn < 3; ++tn) {
    int ch = n0 + wn * 48 + tn * 16 + lr;
#pragma unroll
    for (int tm = 0; tm < 2; ++tm) {
      int mb = m0 + wm * 32 + tm * 16 + half * 8;   // D: VGPR i -> M=i+8*half
      v8h ov;
#pragma unroll
      for (int i = 0; i < 8; ++i) ov[i] = (_Float16)(acc[tm][tn][i] * alpha);
      *(v8h*)&y1[(size_t)ch * kM + mb] = ov;        // 16B contiguous store
    }
  }
}

// ---------------------------------------------------------------------------
// Per-channel mean/var inputs: deterministic fixed-order tree reduction
// ---------------------------------------------------------------------------
__global__ __launch_bounds__(256)
void channel_stats(const _Float16* __restrict__ y, float* __restrict__ sum,
                   float* __restrict__ sq) {
  __shared__ float rs[256], rq[256];
  const int c = blockIdx.x, t = threadIdx.x;
  const _Float16* p = y + (size_t)c * kM;
  float s = 0.f, q = 0.f;
  for (int i = t; i < kM; i += 256) { float v = (float)p[i]; s += v; q += v * v; }
  rs[t] = s; rq[t] = q; __syncthreads();
  for (int off = 128; off; off >>= 1) {
    if (t < off) { rs[t] += rs[t + off]; rq[t] += rq[t + off]; }
    __syncthreads();
  }
  if (!t) { sum[c] = rs[0]; sq[c] = rq[0]; }
}

__global__ __launch_bounds__(256)
void bn_finalize(const float* __restrict__ sum, const float* __restrict__ sq,
                 const float* __restrict__ g, const float* __restrict__ bta,
                 float* __restrict__ A, float* __restrict__ Bv, int C, float invn) {
  int i = blockIdx.x * 256 + threadIdx.x;
  if (i < C) {
    float mean = sum[i] * invn;
    float var  = fmaxf(sq[i] * invn - mean * mean, 0.f);
    float a    = g[i] * rsqrtf(var + 1e-5f);
    A[i] = a; Bv[i] = bta[i] - mean * a;
  }
}

// ---------------------------------------------------------------------------
// Depthwise 3x3 (pad 1): BN1+ReLU applied once into a 58x58 LDS tile, then
// 9 LDS FMAs per pixel (cuts L2 traffic 9x vs direct). Block = one (c,b).
// ---------------------------------------------------------------------------
__global__ __launch_bounds__(256)
void dwconv(const _Float16* __restrict__ y1, const float* __restrict__ w2t,
            const float* __restrict__ bn1A, const float* __restrict__ bn1B,
            _Float16* __restrict__ y2) {
  __shared__ float tile[58 * 58];
  const int t = threadIdx.x;
  const int c = blockIdx.x % kP;
  const int b = blockIdx.x / kP;
  const size_t base = (size_t)c * kM + (size_t)b * kHW;
  const float A = bn1A[c], Bv = bn1B[c];
  float wk[9];
#pragma unroll
  for (int i = 0; i < 9; ++i) wk[i] = w2t[c * 9 + i];   // already alpha2-scaled
  for (int i = t; i < 58 * 58; i += 256) tile[i] = 0.f; // zero pad border
  __syncthreads();
  for (int p = t; p < kHW; p += 256) {
    int h = p / 56, w = p % 56;
    float f = (float)y1[base + p];
    tile[(h + 1) * 58 + (w + 1)] = fmaxf(f * A + Bv, 0.f);  // BN1 + ReLU
  }
  __syncthreads();
  for (int p = t; p < kHW; p += 256) {
    int h = p / 56, w = p % 56;
    float a = 0.f;
#pragma unroll
    for (int dy = 0; dy < 3; ++dy)
#pragma unroll
      for (int dx = 0; dx < 3; ++dx)
        a += wk[dy * 3 + dx] * tile[(h + dy) * 58 + (w + dx)];
    y2[base + p] = (_Float16)a;
  }
}

// ---------------------------------------------------------------------------
// GEMM2: y3[n][m] = alpha3 * sum_k relu(bn2(y2[k][m])) * t3[n][k]
//   block: 8 waves (4M x 2N), tile 128M x 96N (all of N), K-loop 18x32,
//   double-buffered LDS; A prefetched to regs, B async-copied to LDS.
// ---------------------------------------------------------------------------
__global__ __launch_bounds__(256)
void gemm_project(const _Float16* __restrict__ y2, const _Float16* __restrict__ w3t,
                  const float* __restrict__ stats, const float* __restrict__ bn2A,
                  const float* __restrict__ bn2B, _Float16* __restrict__ y3) {
  __shared__ __attribute__((aligned(16))) _Float16 Al[2][128 * 32];  // [m][k]
  __shared__ __attribute__((aligned(16))) _Float16 Bl[2][96 * 32];   // [n][k]
  const int t  = threadIdx.x;
  const int m0 = blockIdx.x * 128;
  const float alpha = stats[5];
  const int wave = t >> 5, lane = t & 31;
  const int half = lane >> 4, lr = lane & 15;
  const int wm = wave >> 1, wn = wave & 1;   // 4M x 2N waves
  const int ml  = t & 127;                   // this thread's fixed A coords
  const int kk0 = t >> 7;                    // 0 or 1; covers kk = kk0 + 2i

  // ---- stage slab 0
#pragma unroll
  for (int i = 0; i < 16; ++i) {
    int c = kk0 + i * 2;
    float f = (float)y2[(size_t)c * kM + m0 + ml];
    Al[0][ml * 32 + c] = (_Float16)fmaxf(f * bn2A[c] + bn2B[c], 0.f);
  }
#if HAVE_ASYNC_LDS
  for (int idx = t; idx < (96 * 32) / 8; idx += 256) {
    int n = idx >> 2, ch = idx & 3;
    async_cp16(&Bl[0][n * 32 + ch * 8], &w3t[(size_t)n * kP + ch * 8]);
  }
  wait_async_all();
#else
  for (int idx = t; idx < 96 * 32; idx += 256)
    Bl[0][idx] = w3t[(size_t)(idx >> 5) * kP + (idx & 31)];
#endif
  __syncthreads();

  v8f acc[2][3] = {};
#pragma unroll 2
  for (int kb = 0; kb < 18; ++kb) {
    const int cur = kb & 1, nxt = cur ^ 1;

    // ---- issue next-slab loads before touching the matrix pipe
    float pf[16];
    if (kb + 1 < 18) {
#pragma unroll
      for (int i = 0; i < 16; ++i) {
        int c = (kb + 1) * 32 + kk0 + i * 2;
        pf[i] = (float)y2[(size_t)c * kM + m0 + ml];
      }
#if HAVE_ASYNC_LDS
      for (int idx = t; idx < (96 * 32) / 8; idx += 256) {
        int n = idx >> 2, ch = idx & 3;
        async_cp16(&Bl[nxt][n * 32 + ch * 8],
                   &w3t[(size_t)n * kP + (kb + 1) * 32 + ch * 8]);
      }
#else
      for (int idx = t; idx < 96 * 32; idx += 256)
        Bl[nxt][idx] = w3t[(size_t)(idx >> 5) * kP + (kb + 1) * 32 + (idx & 31)];
#endif
    }

    // ---- compute current slab
    v16h af[2], bf[3];
#pragma unroll
    for (int tm = 0; tm < 2; ++tm) {
      int m = wm * 32 + tm * 16 + lr;
      af[tm] = combine8(*(const v8h*)&Al[cur][m * 32 + half * 8],
                        *(const v8h*)&Al[cur][m * 32 + 16 + half * 8]);
    }
#pragma unroll
    for (int tn = 0; tn < 3; ++tn) {
      int n = wn * 48 + tn * 16 + lr;
      bf[tn] = combine8(*(const v8h*)&Bl[cur][n * 32 + half * 8],
                        *(const v8h*)&Bl[cur][n * 32 + 16 + half * 8]);
    }
#pragma unroll
    for (int tm = 0; tm < 2; ++tm)
#pragma unroll
      for (int tn = 0; tn < 3; ++tn)
        acc[tm][tn] = __builtin_amdgcn_wmma_f32_16x16x32_f16(
            false, af[tm], false, bf[tn], (short)0, acc[tm][tn], false, false);

    // ---- write prefetched A (BN2+ReLU fused) into the other buffer
    if (kb + 1 < 18) {
#pragma unroll
      for (int i = 0; i < 16; ++i) {
        int c = (kb + 1) * 32 + kk0 + i * 2;
        Al[nxt][ml * 32 + (kk0 + i * 2)] =
            (_Float16)fmaxf(pf[i] * bn2A[c] + bn2B[c], 0.f);
      }
    }
    wait_async_all();
    __syncthreads();
  }

#pragma unroll
  for (int tn = 0; tn < 3; ++tn) {
    int ch = wn * 48 + tn * 16 + lr;          // 0..95
#pragma unroll
    for (int tm = 0; tm < 2; ++tm) {
      int mb = m0 + wm * 32 + tm * 16 + half * 8;
      v8h ov;
#pragma unroll
      for (int i = 0; i < 8; ++i) ov[i] = (_Float16)(acc[tm][tn][i] * alpha);
      *(v8h*)&y3[(size_t)ch * kM + mb] = ov;
    }
  }
}

// out = BN3(y3) + x  (NCHW fp32)
__global__ __launch_bounds__(256)
void bn_residual(const _Float16* __restrict__ y3, const float* __restrict__ bn3A,
                 const float* __restrict__ bn3B, const float* __restrict__ x,
                 float* __restrict__ out, int n) {
  int idx = blockIdx.x * 256 + threadIdx.x;
  if (idx < n) {
    int c  = (idx / kHW) % kC;
    int b  = idx / (kC * kHW);
    int hw = idx % kHW;
    float v = (float)y3[(size_t)c * kM + b * kHW + hw];
    out[idx] = v * bn3A[c] + bn3B[c] + x[idx];
  }
}

// ---------------------------------------------------------------------------
extern "C" void kernel_launch(void* const* d_in, const int* in_sizes, int n_in,
                              void* d_out, int out_size, void* d_ws, size_t ws_size,
                              hipStream_t stream) {
  const float* x  = (const float*)d_in[0];
  const float* w1 = (const float*)d_in[1];
  const float* g1 = (const float*)d_in[2];
  const float* bb1 = (const float*)d_in[3];
  const float* w2 = (const float*)d_in[4];
  const float* g2 = (const float*)d_in[5];
  const float* bb2 = (const float*)d_in[6];
  const float* w3 = (const float*)d_in[7];
  const float* g3 = (const float*)d_in[8];
  const float* bb3 = (const float*)d_in[9];
  float* out = (float*)d_out;

  // ---- workspace layout (needs ~252 MB) ----
  char* ws = (char*)d_ws;
  float* stats = (float*)ws;            // [0..5]: delta/alpha for w1,w2,w3
  float* sum1 = stats + 16;             float* sq1 = sum1 + kP;
  float* sum2 = sq1 + kP;               float* sq2 = sum2 + kP;
  float* sum3 = sq2 + kP;               float* sq3 = sum3 + kC;
  float* bn1A = sq3 + kC;               float* bn1B = bn1A + kP;
  float* bn2A = bn1B + kP;              float* bn2B = bn2A + kP;
  float* bn3A = bn2B + kP;              float* bn3B = bn3A + kC;
  float* w2t  = bn3B + kC;                                  // 5184 f32
  _Float16* w1t = (_Float16*)(ws + 65536);                  // 110592 B
  _Float16* w3t = (_Float16*)(ws + 65536 + 131072);         // 110592 B
  constexpr size_t OFF_Y1 = 1ull << 20;
  constexpr size_t SZY_AL = 111ull << 20;   // 576*100352*2 = 115.6MB, 1MB-aligned
  _Float16* y1 = (_Float16*)(ws + OFF_Y1);
  _Float16* y2 = (_Float16*)(ws + OFF_Y1 + SZY_AL);
  _Float16* y3 = (_Float16*)(ws + OFF_Y1 + 2 * SZY_AL);
  (void)in_sizes; (void)n_in; (void)out_size; (void)ws_size;

  const float invn = 1.f / (float)kM;

  // 1) ternarization stats + weight packing
  ternary_stats<<<1, 256, 0, stream>>>(w1, kP * kC, stats + 0);
  ternary_stats<<<1, 256, 0, stream>>>(w2, kP * 9,  stats + 2);
  ternary_stats<<<1, 256, 0, stream>>>(w3, kC * kP, stats + 4);
  pack_tern_f16 <<<(kP * kC + 255) / 256, 256, 0, stream>>>(w1, w1t, kP * kC, stats + 0);
  pack_tern_f16 <<<(kC * kP + 255) / 256, 256, 0, stream>>>(w3, w3t, kC * kP, stats + 4);
  pack_tern_f32s<<<(kP * 9  + 255) / 256, 256, 0, stream>>>(w2, w2t, kP * 9,  stats + 2);

  // 2) expand GEMM (WMMA) -> y1, then BN1 stats
  dim3 grid1(kM / 64, kP / 192);   // 1568 x 3
  gemm_expand<<<grid1, 256, 0, stream>>>(x, w1t, stats, y1);
  channel_stats<<<kP, 256, 0, stream>>>(y1, sum1, sq1);
  bn_finalize<<<3, 256, 0, stream>>>(sum1, sq1, g1, bb1, bn1A, bn1B, kP, invn);

  // 3) depthwise 3x3 (BN1+ReLU fused, LDS-tiled) -> y2, then BN2 stats
  dwconv<<<kP * kBn, 256, 0, stream>>>(y1, w2t, bn1A, bn1B, y2);
  channel_stats<<<kP, 256, 0, stream>>>(y2, sum2, sq2);
  bn_finalize<<<3, 256, 0, stream>>>(sum2, sq2, g2, bb2, bn2A, bn2B, kP, invn);

  // 4) project GEMM (WMMA, BN2+ReLU fused on A-load, double-buffered) -> y3
  gemm_project<<<kM / 128, 256, 0, stream>>>(y2, w3t, stats, bn2A, bn2B, y3);
  channel_stats<<<kC, 256, 0, stream>>>(y3, sum3, sq3);
  bn_finalize<<<1, 256, 0, stream>>>(sum3, sq3, g3, bb3, bn3A, bn3B, kC, invn);

  // 5) BN3 + identity shortcut
  int nOut = kBn * kC * kHW;
  bn_residual<<<(nOut + 255) / 256, 256, 0, stream>>>(y3, bn3A, bn3B, x, out, nOut);
}
